// CrossTransformerLayer_37022618092056
// MI455X (gfx1250) — matrix operand order repaired and checked by reading, not compiled, not run on hardware
//
#include <hip/hip_runtime.h>
#include <stddef.h>
#include <stdint.h>

// ---------------------------------------------------------------------------
// CrossTransformerLayer for MI455X (gfx1250, wave32, WMMA + TDM).
// Edge tensor (2,256,256,128) dominates -> edge intermediates stored bf16
// (HBM 23.3 TB/s is the binding resource), all 128x128 linears run through
// v_wmma_f32_16x16x32_bf16 with fp32 accumulation. W is pre-packed in LDS in
// the exact B-fragment layout so fragment loads are contiguous ds_load_b128;
// the X tile is bank-padded (288B row stride). For bf16 inputs the X tile is
// staged by the Tensor Data Mover (tensor_load_to_lds) with hardware row
// padding (pad_interval=64 DWORDs, pad_amount=8 DWORDs), tracked by TENSORcnt.
// ---------------------------------------------------------------------------

typedef __bf16 bf16;
typedef __attribute__((ext_vector_type(16))) __bf16 v16bf;
typedef __attribute__((ext_vector_type(8)))  __bf16 bf16x8;
typedef __attribute__((ext_vector_type(4)))  __bf16 bf16x4;
typedef __attribute__((ext_vector_type(8)))  float  v8f;
typedef __attribute__((ext_vector_type(4)))  unsigned int u32x4;
typedef __attribute__((ext_vector_type(8)))  int i32x8;
typedef __attribute__((ext_vector_type(4)))  int i32x4;

#if __has_builtin(__builtin_amdgcn_tensor_load_to_lds) && \
    __has_builtin(__builtin_amdgcn_s_wait_tensorcnt)
#define HAVE_TDM 1
#else
#define HAVE_TDM 0
#endif

#define DEVINL __device__ __forceinline__

constexpr int LDA = 144;  // padded bf16 elems per X-tile row (288B, 16B-aligned)

// ---- WMMA fragment loaders ------------------------------------------------
// A (16x32, 16-bit): lane<16: row=lane, K = {kh..kh+7, 16+kh..23+kh}, kh=0
//                    lane>=16: same row, kh=8. Each half contiguous 16B.
DEVINL v16bf load_A_frag(const bf16* rowbase, int lane) {
  int row = lane & 15;
  int kh  = (lane & 16) ? 8 : 0;
  const bf16* r = rowbase + row * LDA + kh;
  bf16x8 lo = *(const bf16x8*)(r);
  bf16x8 hi = *(const bf16x8*)(r + 16);
  return __builtin_shufflevector(lo, hi, 0, 1, 2, 3, 4, 5, 6, 7,
                                 8, 9, 10, 11, 12, 13, 14, 15);
}

// B fragments are pre-packed: one contiguous 32B chunk per (ks,nt,lane).
DEVINL v16bf load_B_frag(const bf16* sWp, int ks, int nt, int lane) {
  return *(const v16bf*)(sWp + (((ks * 8 + nt) * 32) + lane) * 16);
}

DEVINL float leaky05(float x) { return x > 0.f ? x : 0.05f * x; }

// ---- Generic fused GEMM: Y[M,128] = act(X[M,128] @ W[128,128] + b [+ R]) --
// 256 threads = 8 waves; each wave -> 16 rows x 128 cols (8 WMMA N-tiles).
template <typename TX, typename TY, int ACT, bool RES>
__global__ __launch_bounds__(256) void k_gemm128(
    const TX* __restrict__ X, int ldx,
    const float* __restrict__ W, int ldw,
    const float* __restrict__ bias,
    const TY* resid, TY* Y, int ldy, int M) {
  __shared__ __align__(32) bf16 sX[128 * LDA];
  __shared__ __align__(32) bf16 sWp[128 * 128];  // packed B-fragment layout
  const int tid  = threadIdx.x;
  const int lane = tid & 31;
  const int wave = tid >> 5;
  const int row0 = blockIdx.x * 128;

  // Pack W[k][n] (fp32) into B-fragment layout (bf16). Reads coalesced.
  for (int i = tid; i < 128 * 128; i += 256) {
    int k = i >> 7, n = i & 127;
    int ks = k >> 5, kr = k & 31;
    int hi = (kr >> 3) & 1, e = (kr & 7) | ((kr >> 4) << 3);
    int nt = n >> 4, ln = n & 15;
    int pl = hi * 16 + ln;
    sWp[(((ks * 8 + nt) * 32) + pl) * 16 + e] = (bf16)W[(size_t)k * ldw + n];
  }

  if constexpr (sizeof(TX) == 2 && HAVE_TDM) {
    // --- TDM path: DMA the bf16 tile global->LDS with hardware row padding.
    // Rows of 128 x 2B = 256B = 64 DWORDs; pad 8 DWORDs (32B) after each row
    // -> 288B LDS row stride == LDA. OOB rows would read zero (M%128==0 here).
    if (wave == 0) {
      unsigned lds_base = (unsigned)(uintptr_t)&sX[0];
      unsigned long long ga =
          (unsigned long long)(uintptr_t)(X + (size_t)row0 * ldx);
      u32x4 g0;
      g0[0] = 1u;                                   // count=1, user descriptor
      g0[1] = lds_base;                             // lds_addr (bytes)
      g0[2] = (unsigned)(ga & 0xffffffffu);         // global_addr[31:0]
      g0[3] = (unsigned)((ga >> 32) & 0x1ffffffu) | (2u << 30);  // type=2
      i32x8 g1;
      g1[0] = (int)((1u << 16) |   // data_size = 2 bytes
                    (1u << 20) |   // pad_enable
                    (5u << 22) |   // pad_interval: 64 DWORDs
                    (7u << 25));   // pad_amount: 8 DWORDs
      g1[1] = (int)(128u << 16);   // tensor_dim0 = 128 (low half)
      g1[2] = (int)(128u << 16);   // tensor_dim0 hi=0 | tensor_dim1 lo = 128
      g1[3] = (int)(128u << 16);   // tensor_dim1 hi=0 | tile_dim0 = 128
      g1[4] = (int)128;            // tile_dim1 = 128, tile_dim2 = 0
      g1[5] = (int)(unsigned)ldx;  // tensor_dim0_stride (data_size units)
      g1[6] = 0;                   // dim0_stride hi | tensor_dim1_stride lo
      g1[7] = 0;                   // tensor_dim1_stride hi
      i32x4 g2 = {0, 0, 0, 0};     // group 2 unused (2-D tensor)
      i32x4 g3 = {0, 0, 0, 0};     // group 3 unused
      i32x8 g4 = {0, 0, 0, 0, 0, 0, 0, 0};  // extra group (clang-23 form)
      __builtin_amdgcn_tensor_load_to_lds(g0, g1, g2, g3, g4, 0);
      __builtin_amdgcn_s_wait_tensorcnt(0);
    }
  } else {
    // --- Manual path (fp32 input needs conversion), 4 elems per iteration.
    for (int i4 = tid; i4 < 128 * 32; i4 += 256) {
      int r = i4 >> 5, c4 = (i4 & 31) * 4;
      int gr = row0 + r;
      bf16x4 v;
      if (gr < M) {
        const TX* xp = X + (size_t)gr * ldx + c4;
#pragma unroll
        for (int u = 0; u < 4; ++u) v[u] = (bf16)(float)xp[u];
      } else {
#pragma unroll
        for (int u = 0; u < 4; ++u) v[u] = (bf16)0.f;
      }
      *(bf16x4*)&sX[r * LDA + c4] = v;
    }
  }
  __syncthreads();

  const int wrow = wave * 16;
  v8f acc[8];
#pragma unroll
  for (int t = 0; t < 8; ++t)
#pragma unroll
    for (int v = 0; v < 8; ++v) acc[t][v] = 0.f;

#pragma unroll
  for (int ks = 0; ks < 4; ++ks) {
    v16bf a = load_A_frag(&sX[wrow * LDA + ks * 32], lane);
#pragma unroll
    for (int nt = 0; nt < 8; ++nt) {
      v16bf b = load_B_frag(sWp, ks, nt, lane);
      acc[nt] = __builtin_amdgcn_wmma_f32_16x16x32_bf16(
          false, a, false, b, (short)0, acc[nt], false, false);
    }
  }

  // D layout: lanes 0-15 col=lane, VGPR v -> M=v ; lanes 16-31 -> M=8+v
  const int colb = lane & 15;
  const int rofs = (lane & 16) ? 8 : 0;
#pragma unroll
  for (int nt = 0; nt < 8; ++nt) {
#pragma unroll
    for (int v = 0; v < 8; ++v) {
      int gr = row0 + wrow + rofs + v;
      if (gr < M) {
        int gc = nt * 16 + colb;
        float val = acc[nt][v] + (bias ? bias[gc] : 0.f);
        if (RES) val += (float)resid[(size_t)gr * ldy + gc];
        if (ACT == 1) val = leaky05(val);
        Y[(size_t)gr * ldy + gc] = (TY)val;
      }
    }
  }
}

// ---- Skip + LayerNorm (one wave32 per 128-wide row) -----------------------
template <typename TX, typename TR, typename TY, bool AFFINE, int ACT>
__global__ __launch_bounds__(256) void k_skip_ln(
    const TX* __restrict__ x, const TR* __restrict__ res,
    const float* __restrict__ g, const float* __restrict__ bta,
    TY* __restrict__ y, int rows) {
  int lane = threadIdx.x & 31;
  int wave = threadIdx.x >> 5;
  int row  = blockIdx.x * 8 + wave;
  if (row >= rows) return;
  const TX* xr = x + (size_t)row * 128;
  const TR* rr = res + (size_t)row * 128;
  float v[4];
  float s = 0.f;
#pragma unroll
  for (int i = 0; i < 4; ++i) {
    int c = lane * 4 + i;
    float t = (float)xr[c] + (float)rr[c];
    v[i] = t;
    s += t;
  }
#pragma unroll
  for (int o = 16; o >= 1; o >>= 1) s += __shfl_xor(s, o, 32);
  float mean = s * (1.f / 128.f);
  float q = 0.f;
#pragma unroll
  for (int i = 0; i < 4; ++i) { float d = v[i] - mean; q += d * d; }
#pragma unroll
  for (int o = 16; o >= 1; o >>= 1) q += __shfl_xor(q, o, 32);
  float rstd = rsqrtf(q * (1.f / 128.f) + 1e-5f);
#pragma unroll
  for (int i = 0; i < 4; ++i) {
    int c = lane * 4 + i;
    float t = (v[i] - mean) * rstd;
    if (AFFINE) t = t * g[c] + bta[c];
    if (ACT == 1) t = leaky05(t);
    y[(size_t)row * 128 + c] = (TY)t;
  }
}

// ---- FiLM per-batch vectors, node-sized (scalar, generic) -----------------
template <typename TX, typename TY>
__global__ __launch_bounds__(256) void k_film_vec(
    const TX* __restrict__ x, const float* __restrict__ mul,
    const float* __restrict__ add, TY* __restrict__ y,
    size_t rows, int rowsPerBatch) {
  size_t idx = (size_t)blockIdx.x * 256 + threadIdx.x;
  if (idx >= rows * 128) return;
  size_t r = idx >> 7;
  int c = (int)(idx & 127);
  int b = (int)(r / (size_t)rowsPerBatch);
  float a = (float)x[idx];
  y[idx] = (TY)(a * (mul[b * 128 + c] + 1.f) + add[b * 128 + c]);
}

// ---- FiLM per-batch vectors, edge-sized, 8-wide (f32 -> bf16) -------------
__global__ __launch_bounds__(256) void k_film_vec_f2b_v8(
    const float* __restrict__ x, const float* __restrict__ mul,
    const float* __restrict__ add, bf16* __restrict__ y,
    size_t nvec, int rowsPerBatch) {
  size_t idx = (size_t)blockIdx.x * 256 + threadIdx.x;
  if (idx >= nvec) return;
  size_t r = idx >> 4;
  int c0 = (int)(idx & 15) * 8;
  int b = (int)(r / (size_t)rowsPerBatch);
  const float* xp = x + r * 128 + c0;
  const float* m = mul + b * 128 + c0;
  const float* a = add + b * 128 + c0;
  float4 lo = ((const float4*)xp)[0];
  float4 hi = ((const float4*)xp)[1];
  float xv[8] = {lo.x, lo.y, lo.z, lo.w, hi.x, hi.y, hi.z, hi.w};
  bf16x8 o;
#pragma unroll
  for (int u = 0; u < 8; ++u) o[u] = (bf16)(xv[u] * (m[u] + 1.f) + a[u]);
  *(bf16x8*)(y + r * 128 + c0) = o;
}

// ---- FiLM per-batch vectors, edge-sized, 8-wide (bf16 -> bf16) ------------
__global__ __launch_bounds__(256) void k_film_vec_b2b_v8(
    const bf16* __restrict__ x, const float* __restrict__ mul,
    const float* __restrict__ add, bf16* __restrict__ y,
    size_t nvec, int rowsPerBatch) {
  size_t idx = (size_t)blockIdx.x * 256 + threadIdx.x;
  if (idx >= nvec) return;
  size_t r = idx >> 4;
  int c0 = (int)(idx & 15) * 8;
  int b = (int)(r / (size_t)rowsPerBatch);
  bf16x8 xv = *(const bf16x8*)(x + r * 128 + c0);
  const float* m = mul + b * 128 + c0;
  const float* a = add + b * 128 + c0;
  bf16x8 o;
#pragma unroll
  for (int u = 0; u < 8; ++u)
    o[u] = (bf16)((float)xv[u] * (m[u] + 1.f) + a[u]);
  *(bf16x8*)(y + r * 128 + c0) = o;
}

// ---- FiLM full: y = a*(m+1) + ad, 8-wide bf16 -----------------------------
__global__ __launch_bounds__(256) void k_film_full_v8(
    const bf16* __restrict__ A, const bf16* __restrict__ Mu,
    const bf16* __restrict__ Ad, bf16* __restrict__ Y, size_t nvec) {
  size_t idx = (size_t)blockIdx.x * 256 + threadIdx.x;
  if (idx >= nvec) return;
  size_t o = idx * 8;
  bf16x8 a = *(const bf16x8*)(A + o);
  bf16x8 m = *(const bf16x8*)(Mu + o);
  bf16x8 d = *(const bf16x8*)(Ad + o);
  bf16x8 y;
#pragma unroll
  for (int u = 0; u < 8; ++u)
    y[u] = (bf16)((float)a[u] * ((float)m[u] + 1.f) + (float)d[u]);
  *(bf16x8*)(Y + o) = y;
}

// ---- QK outer product, 8-wide: AC[bi,j,d] = Q[bi,d]*K[b,j,d]*scale --------
__global__ __launch_bounds__(256) void k_outer_v8(
    const float* __restrict__ Q, const float* __restrict__ K,
    bf16* __restrict__ Y, float scale, size_t nvec) {
  size_t idx = (size_t)blockIdx.x * 256 + threadIdx.x;
  if (idx >= nvec) return;
  int d0 = (int)(idx & 15) * 8;
  size_t j  = (idx >> 4) & 255;
  size_t bi = idx >> 12;  // b*256+i
  int b = (int)(bi >> 8);
  const float* qp = Q + bi * 128 + d0;
  const float* kp = K + ((size_t)b * 256 + j) * 128 + d0;
  float4 qlo = ((const float4*)qp)[0], qhi = ((const float4*)qp)[1];
  float4 klo = ((const float4*)kp)[0], khi = ((const float4*)kp)[1];
  float qv[8] = {qlo.x, qlo.y, qlo.z, qlo.w, qhi.x, qhi.y, qhi.z, qhi.w};
  float kv[8] = {klo.x, klo.y, klo.z, klo.w, khi.x, khi.y, khi.z, khi.w};
  bf16x8 y;
#pragma unroll
  for (int u = 0; u < 8; ++u) y[u] = (bf16)(qv[u] * kv[u] * scale);
  *(bf16x8*)(Y + idx * 8) = y;
}

// ---- Per-head sum over head_dim=16: L[bi,h,j] = sum_t A[bi,j,h*16+t] ------
__global__ __launch_bounds__(256) void k_headsum(
    const bf16* __restrict__ A, float* __restrict__ L, int total) {
  int idx = blockIdx.x * 256 + threadIdx.x;
  if (idx >= total) return;  // total = 2*256*256*8
  int h  = idx & 7;
  int j  = (idx >> 3) & 255;
  int bi = idx >> 11;
  const bf16* p = A + (((size_t)bi * 256 + j) * 128 + h * 16);
  bf16x8 a = *(const bf16x8*)(p);
  bf16x8 b = *(const bf16x8*)(p + 8);
  float s = 0.f;
#pragma unroll
  for (int t = 0; t < 8; ++t) s += (float)a[t] + (float)b[t];
  L[((size_t)bi * 8 + h) * 256 + j] = s;
}

// ---- Softmax over 256 (one wave per row, in place) ------------------------
__global__ __launch_bounds__(256) void k_softmax256(float* L, int rows) {
  int lane = threadIdx.x & 31;
  int wave = threadIdx.x >> 5;
  int row  = blockIdx.x * 8 + wave;
  if (row >= rows) return;
  float* p = L + (size_t)row * 256;
  float v[8];
  float mx = -3.4e38f;
#pragma unroll
  for (int i = 0; i < 8; ++i) { v[i] = p[lane * 8 + i]; mx = fmaxf(mx, v[i]); }
#pragma unroll
  for (int o = 16; o >= 1; o >>= 1) mx = fmaxf(mx, __shfl_xor(mx, o, 32));
  float s = 0.f;
#pragma unroll
  for (int i = 0; i < 8; ++i) { v[i] = __expf(v[i] - mx); s += v[i]; }
#pragma unroll
  for (int o = 16; o >= 1; o >>= 1) s += __shfl_xor(s, o, 32);
  float inv = 1.f / s;
#pragma unroll
  for (int i = 0; i < 8; ++i) p[lane * 8 + i] = v[i] * inv;
}

// ---- Attention aggregation: out[bi,d] = sum_j S[bi,h(d),j]*V[b,j,d] -------
__global__ __launch_bounds__(128) void k_attn_agg(
    const float* __restrict__ S, const float* __restrict__ V,
    float* __restrict__ out) {
  int d  = threadIdx.x;   // 0..127
  int bi = blockIdx.x;    // 0..511
  int b  = bi >> 8;
  int h  = d >> 4;
  const float* srow = S + ((size_t)bi * 8 + h) * 256;
  const float* vb   = V + (size_t)b * 256 * 128 + d;
  float acc = 0.f;
  for (int j = 0; j < 256; ++j) acc += srow[j] * vb[(size_t)j * 128];
  out[(size_t)bi * 128 + d] = acc;
}

// ---- Tiny linear for time vectors (M<=2 rows) -----------------------------
template <int ACT>
__global__ __launch_bounds__(128) void k_lin_small(
    const float* __restrict__ X, const float* __restrict__ W, int ldw,
    const float* __restrict__ bias, float* __restrict__ Y, int M) {
  int c = threadIdx.x;
  int m = blockIdx.x;
  if (m >= M) return;
  float acc = bias ? bias[c] : 0.f;
  for (int k = 0; k < 128; ++k) acc += X[m * 128 + k] * W[k * ldw + c];
  if (ACT == 1) acc = leaky05(acc);
  Y[m * 128 + c] = acc;
}

// ---------------------------------------------------------------------------
// Host-side orchestration.
// Input leaf order assumption: setup_inputs() dict insertion order, DFS:
//  0 classes  1 params  2 edges  3 times, then weights:
//  4,5 t_in.{w,b}
//  6-13 tc1.c film {mul.w,mul.b,add.w,add.b,o1.w,o1.b,o2.w,o2.b}
//  14-21 tc1.p  22-29 tc1.e
//  30-35 s1 {c.g,c.b,p.g,p.b,e.g,e.b}
//  36,37 attn.cqkv  38,39 attn.pqkv  40-47 attn.fce  48-55 attn.fpe
//  56,57 attn.cout  58,59 attn.pout  60,61 attn.eout
//  62-67 s2   68,69 t_out   70-77 tc2.c  78-85 tc2.p  86-93 tc2.e
//  94-99 s3   100-103 mlp_c 104-107 mlp_p 108-111 mlp_e  112-117 s4
// ---------------------------------------------------------------------------

extern "C" void kernel_launch(void* const* d_in, const int* in_sizes, int n_in,
                              void* d_out, int out_size, void* d_ws,
                              size_t ws_size, hipStream_t stream) {
  (void)in_sizes; (void)n_in; (void)out_size;

  const float* classes = (const float*)d_in[0];
  const float* params  = (const float*)d_in[1];
  const float* edges   = (const float*)d_in[2];
  const float* times   = (const float*)d_in[3];
#define WP(i) ((const float*)d_in[(i)])

  const size_t EDGE  = (size_t)2 * 256 * 256 * 128;  // 16,777,216
  const int    MROWE = 131072;                       // edge rows (B*N*N)
  const int    MROWN = 512;                          // node rows (B*N)

  // ---- outputs (fp32): c, p, e, nt -----------------------------------------
  float* out_c  = (float*)d_out;
  float* out_p  = out_c + 65536;
  float* out_e  = out_p + 65536;
  float* out_nt = out_e + EDGE;
  // out_e region (64 MiB fp32) doubles as 2 bf16 edge scratch buffers; both
  // are dead before the final s4 write into out_e.
  bf16* EB3 = (bf16*)out_e;
  bf16* EB4 = EB3 + EDGE;

  // ---- workspace carve -----------------------------------------------------
  char*  wsp = (char*)d_ws;
  size_t off = 0;
  auto carve = [&](size_t bytes) -> void* {
    void* p = wsp + off;
    off += (bytes + 255) & ~(size_t)255;
    return p;
  };
  bf16* EB0 = (bf16*)carve(EDGE * 2);
  bf16* EB1 = (bf16*)carve(EDGE * 2);
  bf16* EB2 = (bf16*)carve(EDGE * 2);
  float* NB[14];
  for (int i = 0; i < 14; ++i) NB[i] = (float*)carve((size_t)65536 * 4);
  float* LG0 = (float*)carve((size_t)2 * 256 * 8 * 256 * 4);
  float* LG1 = (float*)carve((size_t)2 * 256 * 8 * 256 * 4);
  float* t0 = (float*)carve(256 * 4);
  float* t1 = (float*)carve(256 * 4);
  float* vm = (float*)carve(256 * 4);
  float* va = (float*)carve(256 * 4);
  if (off > ws_size) return;  // need ~115 MB of scratch

  const dim3 TB(256);
  const dim3 GN((MROWN + 127) / 128);       // node GEMM blocks (4)
  const dim3 GE((MROWE + 127) / 128);       // edge GEMM blocks (1024)
  const dim3 GEV((unsigned)(EDGE / 8 / 256));  // edge 8-wide elementwise (8192)
  const dim3 GNW(65536 / 256);              // node elementwise (256)
  const dim3 GLN_E(MROWE / 8), GLN_N(MROWN / 8);
  const size_t NVEC = EDGE / 8;
  const float iscale = 0.088388347648318447f;  // 1/sqrt(128)

  // ================= t_in ====================================================
  k_lin_small<1><<<2, 128, 0, stream>>>(times, WP(4), 128, WP(5), t0, 2);

  // ================= tc1 (time conditioning) ================================
  // c
  k_lin_small<0><<<2, 128, 0, stream>>>(t0, WP(6), 128, WP(7), vm, 2);
  k_lin_small<0><<<2, 128, 0, stream>>>(t0, WP(8), 128, WP(9), va, 2);
  k_film_vec<float, float><<<GNW, TB, 0, stream>>>(classes, vm, va, NB[0], MROWN, 256);
  k_gemm128<float, float, 1, false><<<GN, TB, 0, stream>>>(NB[0], 128, WP(10), 128, WP(11), nullptr, NB[1], 128, MROWN);
  k_gemm128<float, float, 1, false><<<GN, TB, 0, stream>>>(NB[1], 128, WP(12), 128, WP(13), nullptr, NB[0], 128, MROWN);
  // p
  k_lin_small<0><<<2, 128, 0, stream>>>(t0, WP(14), 128, WP(15), vm, 2);
  k_lin_small<0><<<2, 128, 0, stream>>>(t0, WP(16), 128, WP(17), va, 2);
  k_film_vec<float, float><<<GNW, TB, 0, stream>>>(params, vm, va, NB[2], MROWN, 256);
  k_gemm128<float, float, 1, false><<<GN, TB, 0, stream>>>(NB[2], 128, WP(18), 128, WP(19), nullptr, NB[3], 128, MROWN);
  k_gemm128<float, float, 1, false><<<GN, TB, 0, stream>>>(NB[3], 128, WP(20), 128, WP(21), nullptr, NB[2], 128, MROWN);
  // e
  k_lin_small<0><<<2, 128, 0, stream>>>(t0, WP(22), 128, WP(23), vm, 2);
  k_lin_small<0><<<2, 128, 0, stream>>>(t0, WP(24), 128, WP(25), va, 2);
  k_film_vec_f2b_v8<<<GEV, TB, 0, stream>>>(edges, vm, va, EB0, NVEC, 65536);
  k_gemm128<bf16, bf16, 1, false><<<GE, TB, 0, stream>>>(EB0, 128, WP(26), 128, WP(27), nullptr, EB1, 128, MROWE);
  k_gemm128<bf16, bf16, 1, false><<<GE, TB, 0, stream>>>(EB1, 128, WP(28), 128, WP(29), nullptr, EB0, 128, MROWE);

  // ================= s1 =====================================================
  k_skip_ln<float, float, float, true, 0><<<GLN_N, TB, 0, stream>>>(NB[0], classes, WP(30), WP(31), NB[4], MROWN);
  k_skip_ln<float, float, float, true, 1><<<GLN_N, TB, 0, stream>>>(NB[2], params, WP(32), WP(33), NB[5], MROWN);
  k_skip_ln<bf16, float, bf16, true, 0><<<GLN_E, TB, 0, stream>>>(EB0, edges, WP(34), WP(35), EB1, MROWE);
  // oc=NB4  op=NB5  oe=EB1

  // ================= cross attention ========================================
  // QKV (weights (128,384), column slices)
  k_gemm128<float, float, 0, false><<<GN, TB, 0, stream>>>(NB[4], 128, WP(36) + 0,   384, WP(37) + 0,   nullptr, NB[6], 128, MROWN);  // Qc
  k_gemm128<float, float, 0, false><<<GN, TB, 0, stream>>>(NB[4], 128, WP(36) + 128, 384, WP(37) + 128, nullptr, NB[7], 128, MROWN);  // Kc
  k_gemm128<float, float, 0, false><<<GN, TB, 0, stream>>>(NB[4], 128, WP(36) + 256, 384, WP(37) + 256, nullptr, NB[8], 128, MROWN);  // Vc
  k_gemm128<float, float, 0, false><<<GN, TB, 0, stream>>>(NB[5], 128, WP(38) + 0,   384, WP(39) + 0,   nullptr, NB[9],  128, MROWN); // Qp
  k_gemm128<float, float, 0, false><<<GN, TB, 0, stream>>>(NB[5], 128, WP(38) + 128, 384, WP(39) + 128, nullptr, NB[10], 128, MROWN); // Kp
  k_gemm128<float, float, 0, false><<<GN, TB, 0, stream>>>(NB[5], 128, WP(38) + 256, 384, WP(39) + 256, nullptr, NB[11], 128, MROWN); // Vp

  // ac branch (FiLM by edges: fce)
  k_outer_v8<<<GEV, TB, 0, stream>>>(NB[6], NB[7], EB0, iscale, NVEC);  // AC
  k_gemm128<bf16, bf16, 0, false><<<GE, TB, 0, stream>>>(EB1, 128, WP(40), 128, WP(41), nullptr, EB2, 128, MROWE);  // mul
  k_gemm128<bf16, bf16, 0, false><<<GE, TB, 0, stream>>>(EB1, 128, WP(42), 128, WP(43), nullptr, EB3, 128, MROWE);  // add
  k_film_full_v8<<<GEV, TB, 0, stream>>>(EB0, EB2, EB3, EB2, NVEC);
  k_gemm128<bf16, bf16, 1, false><<<GE, TB, 0, stream>>>(EB2, 128, WP(44), 128, WP(45), nullptr, EB3, 128, MROWE);
  k_gemm128<bf16, bf16, 1, false><<<GE, TB, 0, stream>>>(EB3, 128, WP(46), 128, WP(47), nullptr, EB0, 128, MROWE);  // ACF

  // ap branch (fpe)
  k_outer_v8<<<GEV, TB, 0, stream>>>(NB[9], NB[10], EB2, iscale, NVEC);  // AP
  k_gemm128<bf16, bf16, 0, false><<<GE, TB, 0, stream>>>(EB1, 128, WP(48), 128, WP(49), nullptr, EB3, 128, MROWE);
  k_gemm128<bf16, bf16, 0, false><<<GE, TB, 0, stream>>>(EB1, 128, WP(50), 128, WP(51), nullptr, EB4, 128, MROWE);
  k_film_full_v8<<<GEV, TB, 0, stream>>>(EB2, EB3, EB4, EB3, NVEC);
  k_gemm128<bf16, bf16, 1, false><<<GE, TB, 0, stream>>>(EB3, 128, WP(52), 128, WP(53), nullptr, EB4, 128, MROWE);
  k_gemm128<bf16, bf16, 1, false><<<GE, TB, 0, stream>>>(EB4, 128, WP(54), 128, WP(55), nullptr, EB2, 128, MROWE);  // APF

  // attention scores + softmax
  k_headsum<<<4096, TB, 0, stream>>>(EB0, LG0, 2 * 256 * 256 * 8);  // sc logits
  k_headsum<<<4096, TB, 0, stream>>>(EB2, LG1, 2 * 256 * 256 * 8);  // sp logits
  k_softmax256<<<512, TB, 0, stream>>>(LG0, 4096);
  k_softmax256<<<512, TB, 0, stream>>>(LG1, 4096);

  // cross aggregation: wc = sp @ Vc ; wp = sc @ Vp
  k_attn_agg<<<512, 128, 0, stream>>>(LG1, NB[8],  NB[12]);
  k_attn_agg<<<512, 128, 0, stream>>>(LG0, NB[11], NB[13]);
  k_gemm128<float, float, 0, false><<<GN, TB, 0, stream>>>(NB[12], 128, WP(56), 128, WP(57), nullptr, NB[6], 128, MROWN);  // c_attn
  k_gemm128<float, float, 0, false><<<GN, TB, 0, stream>>>(NB[13], 128, WP(58), 128, WP(59), nullptr, NB[7], 128, MROWN);  // p_attn

  // eout: concat(ACF,APF) @ W(256,128) + b  -> two accumulated GEMMs
  k_gemm128<bf16, bf16, 0, false><<<GE, TB, 0, stream>>>(EB0, 128, WP(60), 128, nullptr, nullptr, EB3, 128, MROWE);
  k_gemm128<bf16, bf16, 0, true><<<GE, TB, 0, stream>>>(EB2, 128, WP(60) + 128 * 128, 128, WP(61), EB3, EB3, 128, MROWE);

  // ================= s2 =====================================================
  k_skip_ln<float, float, float, true, 0><<<GLN_N, TB, 0, stream>>>(NB[6], NB[4], WP(62), WP(63), NB[0], MROWN);
  k_skip_ln<float, float, float, true, 1><<<GLN_N, TB, 0, stream>>>(NB[7], NB[5], WP(64), WP(65), NB[1], MROWN);
  k_skip_ln<bf16, bf16, bf16, true, 0><<<GLN_E, TB, 0, stream>>>(EB3, EB1, WP(66), WP(67), EB4, MROWE);
  // oc=NB0 op=NB1 oe=EB4

  // ================= t_out ==================================================
  k_lin_small<1><<<2, 128, 0, stream>>>(t0, WP(68), 128, WP(69), t1, 2);

  // ================= tc2 ====================================================
  k_lin_small<0><<<2, 128, 0, stream>>>(t1, WP(70), 128, WP(71), vm, 2);
  k_lin_small<0><<<2, 128, 0, stream>>>(t1, WP(72), 128, WP(73), va, 2);
  k_film_vec<float, float><<<GNW, TB, 0, stream>>>(NB[0], vm, va, NB[2], MROWN, 256);
  k_gemm128<float, float, 1, false><<<GN, TB, 0, stream>>>(NB[2], 128, WP(74), 128, WP(75), nullptr, NB[3], 128, MROWN);
  k_gemm128<float, float, 1, false><<<GN, TB, 0, stream>>>(NB[3], 128, WP(76), 128, WP(77), nullptr, NB[2], 128, MROWN);
  k_lin_small<0><<<2, 128, 0, stream>>>(t1, WP(78), 128, WP(79), vm, 2);
  k_lin_small<0><<<2, 128, 0, stream>>>(t1, WP(80), 128, WP(81), va, 2);
  k_film_vec<float, float><<<GNW, TB, 0, stream>>>(NB[1], vm, va, NB[4], MROWN, 256);
  k_gemm128<float, float, 1, false><<<GN, TB, 0, stream>>>(NB[4], 128, WP(82), 128, WP(83), nullptr, NB[5], 128, MROWN);
  k_gemm128<float, float, 1, false><<<GN, TB, 0, stream>>>(NB[5], 128, WP(84), 128, WP(85), nullptr, NB[4], 128, MROWN);
  k_lin_small<0><<<2, 128, 0, stream>>>(t1, WP(86), 128, WP(87), vm, 2);
  k_lin_small<0><<<2, 128, 0, stream>>>(t1, WP(88), 128, WP(89), va, 2);
  k_film_vec_b2b_v8<<<GEV, TB, 0, stream>>>(EB4, vm, va, EB0, NVEC, 65536);
  k_gemm128<bf16, bf16, 1, false><<<GE, TB, 0, stream>>>(EB0, 128, WP(90), 128, WP(91), nullptr, EB1, 128, MROWE);
  k_gemm128<bf16, bf16, 1, false><<<GE, TB, 0, stream>>>(EB1, 128, WP(92), 128, WP(93), nullptr, EB0, 128, MROWE);

  // ================= s3 =====================================================
  k_skip_ln<float, float, float, true, 0><<<GLN_N, TB, 0, stream>>>(NB[2], NB[0], WP(94), WP(95), NB[6], MROWN);
  k_skip_ln<float, float, float, true, 1><<<GLN_N, TB, 0, stream>>>(NB[4], NB[1], WP(96), WP(97), NB[7], MROWN);
  k_skip_ln<bf16, bf16, bf16, true, 0><<<GLN_E, TB, 0, stream>>>(EB0, EB4, WP(98), WP(99), EB1, MROWE);
  // oc=NB6 op=NB7 oe=EB1  (EB3/EB4 now dead -> out_e region reusable)

  // ================= MLPs ===================================================
  k_gemm128<float, float, 1, false><<<GN, TB, 0, stream>>>(NB[6], 128, WP(100), 128, WP(101), nullptr, NB[3], 128, MROWN);
  k_gemm128<float, float, 1, false><<<GN, TB, 0, stream>>>(NB[3], 128, WP(102), 128, WP(103), nullptr, NB[2], 128, MROWN);
  k_gemm128<float, float, 1, false><<<GN, TB, 0, stream>>>(NB[7], 128, WP(104), 128, WP(105), nullptr, NB[5], 128, MROWN);
  k_gemm128<float, float, 1, false><<<GN, TB, 0, stream>>>(NB[5], 128, WP(106), 128, WP(107), nullptr, NB[4], 128, MROWN);
  k_gemm128<bf16, bf16, 1, false><<<GE, TB, 0, stream>>>(EB1, 128, WP(108), 128, WP(109), nullptr, EB0, 128, MROWE);
  k_gemm128<bf16, bf16, 1, false><<<GE, TB, 0, stream>>>(EB0, 128, WP(110), 128, WP(111), nullptr, EB2, 128, MROWE);

  // ================= s4 -> outputs ==========================================
  k_skip_ln<float, float, float, true, 0><<<GLN_N, TB, 0, stream>>>(NB[2], NB[6], WP(112), WP(113), out_c, MROWN);
  k_skip_ln<float, float, float, true, 1><<<GLN_N, TB, 0, stream>>>(NB[4], NB[7], WP(114), WP(115), out_p, MROWN);
  k_skip_ln<bf16, bf16, float, true, 0><<<GLN_E, TB, 0, stream>>>(EB2, EB1, WP(116), WP(117), out_e, MROWE);

  // ================= nt = ln_noaffine(t1 + times) ===========================
  k_skip_ln<float, float, float, false, 0><<<1, 256, 0, stream>>>(t1, times, nullptr, nullptr, out_nt, 2);
#undef WP
}
